// Encoder_Model_88862873354911
// MI455X (gfx1250) — matrix operand order repaired and checked by reading, not compiled, not run on hardware
//
#include <hip/hip_runtime.h>
#include <math.h>

#define NODE   30000
#define RELSZ  1000
#define TRIPLE 200000
#define DIM    128
#define OUTD   768
#define BATCH  2048
#define GAMMA_ 3.0f
#define LAMB_  30.0f
#define TAU_   10.0f

#define CSPLIT 4
#define CHUNK  7500           // NODE / CSPLIT (columns per z-block)
#define NTC    59             // ceil(CHUNK / 128)
#define KCH    6              // K chunks of 128 (768 / 128)
#define TOT    (NTC * KCH)    // flattened pipeline iterations

#if __has_builtin(__builtin_amdgcn_tensor_load_to_lds) && __has_builtin(__builtin_amdgcn_s_wait_tensorcnt)
#define HAS_TDM 1
#else
#define HAS_TDM 0
#endif

typedef __attribute__((ext_vector_type(16))) __bf16 v16bf;
typedef __attribute__((ext_vector_type(8)))  __bf16 v8bf;
typedef __attribute__((ext_vector_type(8)))  float  v8f;
typedef __attribute__((ext_vector_type(4)))  unsigned v4u;
typedef __attribute__((ext_vector_type(8)))  int    v8i;
typedef __attribute__((ext_vector_type(4)))  int    v4i;

// ---------------- helpers ----------------
__device__ __forceinline__ unsigned short f2bf(float f) {
  unsigned u = __float_as_uint(f);
  u += 0x7fffu + ((u >> 16) & 1u);           // round-to-nearest-even
  return (unsigned short)(u >> 16);
}
// monotonic float -> unsigned key for atomicMax
__device__ __forceinline__ unsigned fkey(float f) {
  unsigned u = __float_as_uint(f);
  return (u & 0x80000000u) ? ~u : (u | 0x80000000u);
}
__device__ __forceinline__ float funkey(unsigned k) {
  unsigned u = (k & 0x80000000u) ? (k & 0x7fffffffu) : ~k;
  return __uint_as_float(u);
}

// ---------------- utility kernels ----------------
__global__ void k_zero_f32(float* p, int n) {
  int i = blockIdx.x * blockDim.x + threadIdx.x;
  if (i < n) p[i] = 0.0f;
}

__global__ void k_sparse_acc(const int* __restrict__ rows, const int* __restrict__ cols,
                             const float* __restrict__ emb, float* __restrict__ outF,
                             int coloff, float* __restrict__ cnt, int nedge) {
  int idx = blockIdx.x * blockDim.x + threadIdx.x;
  int e = idx >> 2;
  if (e >= nedge) return;
  int q = idx & 3;
  int row = rows[e], col = cols[e];
  const float* src = emb + (size_t)col * DIM + q * 32;
  float* dst = outF + (size_t)row * OUTD + coloff + q * 32;
#pragma unroll
  for (int c = 0; c < 32; ++c) atomicAdd(dst + c, src[c]);
  if (q == 0) atomicAdd(cnt + row, 1.0f);
}

__global__ void k_mean_tanh(float* outF, int coloff, const float* __restrict__ cnt) {
  int i = blockIdx.x * blockDim.x + threadIdx.x;
  if (i >= NODE * DIM) return;
  int row = i >> 7, d = i & 127;
  float* p = outF + (size_t)row * OUTD + coloff + d;
  *p = tanhf(*p / fmaxf(cnt[row], 1.0f));
}

__global__ void k_rel_stats(const float* __restrict__ rel_emb,
                            const float* __restrict__ attn_e, const float* __restrict__ attn_r,
                            float* __restrict__ relnorm, float* __restrict__ dots) {
  int gid = blockIdx.x * blockDim.x + threadIdx.x;
  int wv = gid >> 5, lane = gid & 31;
  if (wv >= RELSZ) return;
  const float* rp = rel_emb + (size_t)wv * DIM;
  float nrm = 0.f, d0 = 0.f, d1 = 0.f, d2 = 0.f, d3 = 0.f;
  for (int d = lane; d < DIM; d += 32) {
    float v = rp[d];
    nrm += v * v;
    d0 += v * attn_e[d];        d1 += v * attn_e[DIM + d];
    d2 += v * attn_r[d];        d3 += v * attn_r[DIM + d];
  }
  for (int s = 16; s; s >>= 1) {
    nrm += __shfl_xor(nrm, s, 32);
    d0 += __shfl_xor(d0, s, 32); d1 += __shfl_xor(d1, s, 32);
    d2 += __shfl_xor(d2, s, 32); d3 += __shfl_xor(d3, s, 32);
  }
  if (lane == 0) {
    relnorm[wv] = sqrtf(nrm);
    dots[wv * 4 + 0] = d0; dots[wv * 4 + 1] = d1;
    dots[wv * 4 + 2] = d2; dots[wv * 4 + 3] = d3;
  }
}

__global__ void k_edge_scale(const int* __restrict__ ridv, const float* __restrict__ r_val,
                             const float* __restrict__ relnorm, const float* __restrict__ dots,
                             float* __restrict__ sE, float* __restrict__ att4) {
  int e = blockIdx.x * blockDim.x + threadIdx.x;
  if (e >= TRIPLE) return;
  int rid = ridv[e];
  float rv = r_val[e];
  float s = rv / fmaxf(rv * relnorm[rid], 1e-12f);
  sE[e] = s;
  att4[e * 4 + 0] = s * dots[rid * 4 + 0];
  att4[e * 4 + 1] = s * dots[rid * 4 + 1];
  att4[e * 4 + 2] = s * dots[rid * 4 + 2];
  att4[e * 4 + 3] = s * dots[rid * 4 + 3];
}

__global__ void k_seg_init(unsigned* __restrict__ segmaxU, float* __restrict__ segsum) {
  int i = blockIdx.x * blockDim.x + threadIdx.x;
  if (i < NODE) { segmaxU[i] = 0u; segsum[i] = 0.0f; }
}

__global__ void k_seg_max(const int* __restrict__ rows, const float* __restrict__ att4,
                          int phase, unsigned* __restrict__ segmaxU) {
  int e = blockIdx.x * blockDim.x + threadIdx.x;
  if (e >= TRIPLE) return;
  atomicMax(segmaxU + rows[e], fkey(att4[e * 4 + phase]));
}

__global__ void k_seg_exp(const int* __restrict__ rows, const float* __restrict__ att4,
                          int phase, const unsigned* __restrict__ segmaxU,
                          float* __restrict__ segsum, float* __restrict__ wbuf) {
  int e = blockIdx.x * blockDim.x + threadIdx.x;
  if (e >= TRIPLE) return;
  int row = rows[e];
  float m = funkey(segmaxU[row]);
  float w = expf(att4[e * 4 + phase] - m);
  wbuf[e] = w;
  atomicAdd(segsum + row, w);
}

__global__ void k_aggregate(const int* __restrict__ adj, const int* __restrict__ ridv,
                            const float* __restrict__ sE, const float* __restrict__ wbuf,
                            const float* __restrict__ rel_emb, float* __restrict__ outF,
                            int inoff, int outoff) {
  int gid = blockIdx.x * blockDim.x + threadIdx.x;
  int e = gid >> 5, lane = gid & 31;
  if (e >= TRIPLE) return;
  int row = adj[e], col = adj[TRIPLE + e], rid = ridv[e];
  float s = sE[e], w = wbuf[e];
  float4 f4 = *reinterpret_cast<const float4*>(outF + (size_t)col * OUTD + inoff + lane * 4);
  float4 t4 = *reinterpret_cast<const float4*>(rel_emb + (size_t)rid * DIM + lane * 4);
  t4.x *= s; t4.y *= s; t4.z *= s; t4.w *= s;
  float dot = f4.x * t4.x + f4.y * t4.y + f4.z * t4.z + f4.w * t4.w;
  for (int sh = 16; sh; sh >>= 1) dot += __shfl_xor(dot, sh, 32);
  float* dst = outF + (size_t)row * OUTD + outoff + lane * 4;
  atomicAdd(dst + 0, w * (f4.x - 2.0f * dot * t4.x));
  atomicAdd(dst + 1, w * (f4.y - 2.0f * dot * t4.y));
  atomicAdd(dst + 2, w * (f4.z - 2.0f * dot * t4.z));
  atomicAdd(dst + 3, w * (f4.w - 2.0f * dot * t4.w));
}

__global__ void k_soft_finish(float* outF, int coloff, const float* __restrict__ segsum) {
  int i = blockIdx.x * blockDim.x + threadIdx.x;
  if (i >= NODE * DIM) return;
  int row = i >> 7, d = i & 127;
  float* p = outF + (size_t)row * OUTD + coloff + d;
  float z = segsum[row];
  float v = (z > 0.0f) ? (*p / z) : 0.0f;
  *p = tanhf(v);
}

__global__ void k_embsq(const float* __restrict__ outF, float* __restrict__ embsq) {
  int gid = blockIdx.x * blockDim.x + threadIdx.x;
  int row = gid >> 5, lane = gid & 31;
  if (row >= NODE) return;
  const float* p = outF + (size_t)row * OUTD;
  float s = 0.f;
  for (int d = lane; d < OUTD; d += 32) { float v = p[d]; s += v * v; }
  for (int sh = 16; sh; sh >>= 1) s += __shfl_xor(s, sh, 32);
  if (lane == 0) embsq[row] = s;
}

__global__ void k_tobf16(const float* __restrict__ src, unsigned short* __restrict__ dst, int n) {
  int i = blockIdx.x * blockDim.x + threadIdx.x;
  if (i < n) dst[i] = f2bf(src[i]);
}

__global__ void k_pos(const int* __restrict__ pairs, const float* __restrict__ outF,
                      float* __restrict__ posb) {
  int gid = blockIdx.x * blockDim.x + threadIdx.x;
  int b = gid >> 5, lane = gid & 31;
  if (b >= BATCH) return;
  int l = pairs[b * 2 + 0], r = pairs[b * 2 + 1];
  const float* pl = outF + (size_t)l * OUTD;
  const float* pr = outF + (size_t)r * OUTD;
  float s = 0.f;
  for (int d = lane; d < OUTD; d += 32) { float df = pl[d] - pr[d]; s += df * df; }
  for (int sh = 16; sh; sh >>= 1) s += __shfl_xor(s, sh, 32);
  if (lane == 0) posb[b] = s;
}

// ---------------- TDM descriptor issue: 128 rows x 128 K bf16 tile -> LDS ----------------
#if HAS_TDM
__device__ __forceinline__ void tdm_issue(const unsigned short* gbase, int colbase, int kch,
                                          unsigned ldsoff) {
  unsigned long long ga = (unsigned long long)(size_t)gbase +
                          ((unsigned long long)colbase * OUTD + (unsigned)kch * 128u) * 2ull;
  unsigned rows = (unsigned)(NODE - colbase);      // remaining rows; OOB rows read as 0
  v4u g0;
  g0.x = 1u;                                        // count=1, user mode
  g0.y = ldsoff;                                    // lds_addr (bytes)
  g0.z = (unsigned)(ga & 0xffffffffull);            // global_addr[31:0]
  g0.w = (unsigned)((ga >> 32) & 0x01ffffffull) | (2u << 30);  // addr[56:32] | type=2
  v8i g1;
  g1[0] = (int)(1u << 16);                          // data_size = 1 -> 2 bytes
  g1[1] = (int)(768u << 16);                        // tensor_dim0[15:0] = 768
  g1[2] = (int)((rows & 0xffffu) << 16);            // tensor_dim0[31:16]=0 | tensor_dim1[15:0]
  g1[3] = (int)(((rows >> 16) & 0xffffu) | (128u << 16)); // tensor_dim1[31:16] | tile_dim0=128
  g1[4] = (int)128;                                 // tile_dim1=128 | tile_dim2=0
  g1[5] = (int)768;                                 // tensor_dim0_stride[31:0] = 768
  g1[6] = 0;                                        // stride hi | tensor_dim1_stride lo
  g1[7] = 0;
  v4i gz = {0, 0, 0, 0};
#if defined(__clang_major__) && (__clang_major__ >= 23)
  v8i gz8 = {0, 0, 0, 0, 0, 0, 0, 0};
  __builtin_amdgcn_tensor_load_to_lds(g0, g1, gz, gz, gz8, 0);
#else
  __builtin_amdgcn_tensor_load_to_lds(g0, g1, gz, gz, 0);
#endif
}
#endif

// PASS 0: per-row {sum, sumsq, max}; PASS 1: per-row sum of exp(ils*(loss-max))
template <int PASS>
__global__ __launch_bounds__(256) void k_gemm(const int* __restrict__ pairs,
                                              const unsigned short* __restrict__ embB,
                                              const float* __restrict__ embsq,
                                              const float* __restrict__ posb,
                                              float* __restrict__ rowsum,
                                              float* __restrict__ rowsq,
                                              unsigned* __restrict__ rowmaxU,
                                              const float* __restrict__ rowmaxF,
                                              const float* __restrict__ rowils,
                                              float* __restrict__ rowsumexp) {
  __shared__ __align__(16) unsigned short A_lds[32 * OUTD];     // 48 KB
  __shared__ __align__(256) unsigned short Bsh[2][128 * 128];   // 2 x 32 KB
  __shared__ int lidx32[32], ridx32[32], aidx32[32];
  __shared__ float asq_s[32], pos_s[32];

  const int tid = threadIdx.x;
  const int wave = tid >> 5, lane = tid & 31;
  const int half = lane >> 4, lm = lane & 15;
  const int jloc = wave * 16 + lm;                 // column within 128-wide tile
  const int b0 = blockIdx.x * 32;
  const int branch = blockIdx.y;
  const int cbase = blockIdx.z * CHUNK;
  const int cend = cbase + CHUNK;

  if (tid < 32) {
    int bi = b0 + tid;
    int li = pairs[bi * 2 + 0], ri = pairs[bi * 2 + 1];
    int ai = branch ? ri : li;
    aidx32[tid] = ai; lidx32[tid] = li; ridx32[tid] = ri;
    asq_s[tid] = embsq[ai];
    pos_s[tid] = posb[bi];
  }
  __syncthreads();

  // stage A panel (32 rows x 768 bf16) into LDS
  for (int it = tid; it < 6144; it += 256) {
    int row = it / 192, p = it % 192;
    ushort4 v = *reinterpret_cast<const ushort4*>(embB + (size_t)aidx32[row] * OUTD + p * 4);
    *reinterpret_cast<ushort4*>(&A_lds[row * OUTD + p * 4]) = v;
  }
  __syncthreads();

  float rmaxr[2][8], ilsr[2][8];
  if (PASS == 1) {
#pragma unroll
    for (int t = 0; t < 2; ++t)
#pragma unroll
      for (int r = 0; r < 8; ++r) {
        int grow = branch * BATCH + b0 + t * 16 + r + 8 * half;
        rmaxr[t][r] = rowmaxF[grow];
        ilsr[t][r] = rowils[grow];
      }
  }

  float s0[2][8], s1[2][8], s2[2][8];
#pragma unroll
  for (int t = 0; t < 2; ++t)
#pragma unroll
    for (int r = 0; r < 8; ++r) { s0[t][r] = 0.f; s1[t][r] = 0.f; s2[t][r] = -3.0e38f; }

  const unsigned short* Ap0 = &A_lds[lm * OUTD + half * 8];
  const unsigned short* Ap1 = &A_lds[(16 + lm) * OUTD + half * 8];
  unsigned boff0 = (unsigned)(size_t)(&Bsh[0][0]);   // LDS byte offsets for TDM descriptors
  unsigned boff1 = (unsigned)(size_t)(&Bsh[1][0]);

#if HAS_TDM
  if (wave == 0) {
    tdm_issue(embB, cbase + 0 * 128, 0, boff0);      // tile 0  (nt=0, kch=0)
    tdm_issue(embB, cbase + 0 * 128, 1, boff1);      // tile 1  (nt=0, kch=1)
  }
#else
  (void)boff0; (void)boff1;
#endif

  v8f c0 = {}, c1 = {};
  for (int it = 0; it < TOT; ++it) {
    const int nt = it / KCH;
    const int kch = it - nt * KCH;
#if HAS_TDM
    if (it + 1 < TOT) __builtin_amdgcn_s_wait_tensorcnt(1);
    else              __builtin_amdgcn_s_wait_tensorcnt(0);
    __syncthreads();                                 // publish Bsh[it&1]
#else
    __syncthreads();
    {   // cooperative fallback copy: 128 rows x 128 ushorts, zero-fill OOB rows
      int colbase = cbase + nt * 128;
      const unsigned short* src0 = embB + (size_t)colbase * OUTD + kch * 128;
      unsigned short* Bb = &Bsh[it & 1][0];
      for (int q = tid; q < 2048; q += 256) {
        int row = q >> 4, seg = q & 15;
        uint4 v = {0u, 0u, 0u, 0u};
        if (colbase + row < NODE)
          v = *reinterpret_cast<const uint4*>(src0 + (size_t)row * OUTD + seg * 8);
        *reinterpret_cast<uint4*>(Bb + row * 128 + seg * 8) = v;
      }
    }
    __syncthreads();
#endif

    const unsigned short* Bb = &Bsh[it & 1][0];
    const int kbase = kch * 128;
#pragma unroll
    for (int k2 = 0; k2 < 4; ++k2) {
      int kk = k2 * 32;
      v8bf blo = *reinterpret_cast<const v8bf*>(Bb + jloc * 128 + kk + half * 8);
      v8bf bhi = *reinterpret_cast<const v8bf*>(Bb + jloc * 128 + kk + 16 + half * 8);
      v16bf b = __builtin_shufflevector(blo, bhi, 0, 1, 2, 3, 4, 5, 6, 7,
                                        8, 9, 10, 11, 12, 13, 14, 15);
      v8bf a0lo = *reinterpret_cast<const v8bf*>(Ap0 + kbase + kk);
      v8bf a0hi = *reinterpret_cast<const v8bf*>(Ap0 + kbase + kk + 16);
      v16bf a0 = __builtin_shufflevector(a0lo, a0hi, 0, 1, 2, 3, 4, 5, 6, 7,
                                         8, 9, 10, 11, 12, 13, 14, 15);
      c0 = __builtin_amdgcn_wmma_f32_16x16x32_bf16(false, a0, false, b,
                                                   (short)0, c0, false, false);
      v8bf a1lo = *reinterpret_cast<const v8bf*>(Ap1 + kbase + kk);
      v8bf a1hi = *reinterpret_cast<const v8bf*>(Ap1 + kbase + kk + 16);
      v16bf a1 = __builtin_shufflevector(a1lo, a1hi, 0, 1, 2, 3, 4, 5, 6, 7,
                                         8, 9, 10, 11, 12, 13, 14, 15);
      c1 = __builtin_amdgcn_wmma_f32_16x16x32_bf16(false, a1, false, b,
                                                   (short)0, c1, false, false);
    }

    if (kch == KCH - 1) {      // K complete for this column tile: consume
      int j = cbase + nt * 128 + jloc;
      int jc = (j < NODE) ? j : (NODE - 1);
      bool valid = (j < cend) && (j < NODE);
      float esq = embsq[jc];
#pragma unroll
      for (int t = 0; t < 2; ++t) {
#pragma unroll
        for (int r = 0; r < 8; ++r) {
          int m = t * 16 + r + 8 * half;
          float dot = (t == 0) ? c0[r] : c1[r];
          float neg = asq_s[m] + esq - 2.0f * dot;
          float fac = 1.0f - (j == lidx32[m] ? 1.0f : 0.0f) - (j == ridx32[m] ? 1.0f : 0.0f);
          float loss = (pos_s[m] - neg + GAMMA_) * fac;
          if (PASS == 0) {
            if (valid) { s0[t][r] += loss; s1[t][r] += loss * loss; s2[t][r] = fmaxf(s2[t][r], loss); }
          } else {
            if (valid) s0[t][r] += expf(ilsr[t][r] * (loss - rmaxr[t][r]));
          }
        }
      }
      c0 = (v8f){};
      c1 = (v8f){};
    }

    __syncthreads();                                 // all waves done reading Bsh[it&1]
#if HAS_TDM
    if (wave == 0 && it + 2 < TOT) {
      int it2 = it + 2;
      int nt2 = it2 / KCH, kch2 = it2 - nt2 * KCH;
      tdm_issue(embB, cbase + nt2 * 128, kch2, (it & 1) ? boff1 : boff0);
    }
#endif
  }

  // reduce the 16 column-lanes of each half-wave, then atomic-accumulate per row
#pragma unroll
  for (int t = 0; t < 2; ++t) {
#pragma unroll
    for (int r = 0; r < 8; ++r) {
      for (int sh = 8; sh; sh >>= 1) {
        s0[t][r] += __shfl_xor(s0[t][r], sh, 16);
        if (PASS == 0) {
          s1[t][r] += __shfl_xor(s1[t][r], sh, 16);
          s2[t][r] = fmaxf(s2[t][r], __shfl_xor(s2[t][r], sh, 16));
        }
      }
    }
  }
  if (lm == 0) {
#pragma unroll
    for (int t = 0; t < 2; ++t)
#pragma unroll
      for (int r = 0; r < 8; ++r) {
        int grow = branch * BATCH + b0 + t * 16 + r + 8 * half;
        if (PASS == 0) {
          atomicAdd(rowsum + grow, s0[t][r]);
          atomicAdd(rowsq + grow, s1[t][r]);
          atomicMax(rowmaxU + grow, fkey(s2[t][r]));
        } else {
          atomicAdd(rowsumexp + grow, s0[t][r]);
        }
      }
  }
}

__global__ void k_rowfin(const float* __restrict__ rowsum, const float* __restrict__ rowsq,
                         const unsigned* __restrict__ rowmaxU, float* __restrict__ rowmu,
                         float* __restrict__ rowils, float* __restrict__ rowmaxF,
                         float* __restrict__ rowsumexp) {
  int i = blockIdx.x * blockDim.x + threadIdx.x;
  if (i >= 2 * BATCH) return;
  float mu = rowsum[i] / (float)NODE;
  float var = rowsq[i] / (float)NODE - mu * mu;
  float sd = fmaxf(sqrtf(fmaxf(var, 0.0f)), 1e-20f);
  rowmu[i] = mu;
  rowils[i] = LAMB_ / sd;
  rowmaxF[i] = funkey(rowmaxU[i]);
  rowsumexp[i] = 0.0f;
}

__global__ void k_rowloss(const float* __restrict__ rowmu, const float* __restrict__ rowils,
                          const float* __restrict__ rowmaxF, const float* __restrict__ rowsumexp,
                          float* __restrict__ accum) {
  int i = blockIdx.x * blockDim.x + threadIdx.x;
  if (i >= 2 * BATCH) return;
  float rl = (rowmaxF[i] - rowmu[i]) * rowils[i] + TAU_ + logf(rowsumexp[i]);
  atomicAdd(accum, rl);
}

__global__ void k_final(const float* __restrict__ accum, float* __restrict__ out) {
  out[0] = accum[0] * (1.0f / (float)BATCH);
}

// ---------------- host launcher ----------------
extern "C" void kernel_launch(void* const* d_in, const int* in_sizes, int n_in,
                              void* d_out, int out_size, void* d_ws, size_t ws_size,
                              hipStream_t stream) {
  (void)in_sizes; (void)n_in; (void)out_size; (void)ws_size;
  const int*   pairs    = (const int*)d_in[0];
  const int*   ent_adj  = (const int*)d_in[1];
  const int*   rel_adj  = (const int*)d_in[2];
  const int*   adj_list = (const int*)d_in[3];
  const int*   r_index  = (const int*)d_in[4];
  const float* r_val    = (const float*)d_in[5];
  const float* ent_emb  = (const float*)d_in[7];
  const float* rel_emb  = (const float*)d_in[8];
  const float* attn_e   = (const float*)d_in[9];
  const float* attn_r   = (const float*)d_in[10];

  char* ws = (char*)d_ws;
  size_t off = 0;
  auto alloc = [&](size_t bytes) -> void* {
    void* p = (void*)(ws + off);
    off = (off + bytes + 255) & ~(size_t)255;
    return p;
  };
  float*          outF     = (float*)alloc(sizeof(float) * (size_t)NODE * OUTD);
  unsigned short* embB     = (unsigned short*)alloc(sizeof(unsigned short) * (size_t)NODE * OUTD);
  float*          cntE     = (float*)alloc(sizeof(float) * NODE);
  float*          cntR     = (float*)alloc(sizeof(float) * NODE);
  float*          relnorm  = (float*)alloc(sizeof(float) * RELSZ);
  float*          dots     = (float*)alloc(sizeof(float) * RELSZ * 4);
  float*          sE       = (float*)alloc(sizeof(float) * TRIPLE);
  float*          att4     = (float*)alloc(sizeof(float) * TRIPLE * 4);
  unsigned*       segmaxU  = (unsigned*)alloc(sizeof(unsigned) * NODE);
  float*          segsum   = (float*)alloc(sizeof(float) * NODE);
  float*          wbuf     = (float*)alloc(sizeof(float) * TRIPLE);
  float*          embsq    = (float*)alloc(sizeof(float) * NODE);
  float*          posb     = (float*)alloc(sizeof(float) * BATCH);
  float*          rowsum   = (float*)alloc(sizeof(float) * 2 * BATCH);
  float*          rowsq    = (float*)alloc(sizeof(float) * 2 * BATCH);
  unsigned*       rowmaxU  = (unsigned*)alloc(sizeof(unsigned) * 2 * BATCH);
  float*          rowmu    = (float*)alloc(sizeof(float) * 2 * BATCH);
  float*          rowils   = (float*)alloc(sizeof(float) * 2 * BATCH);
  float*          rowmaxF  = (float*)alloc(sizeof(float) * 2 * BATCH);
  float*          rowsumexp= (float*)alloc(sizeof(float) * 2 * BATCH);
  float*          accum    = (float*)alloc(sizeof(float) * 16);

  const int TB = 256;
  auto blk = [](long n, int t) { return dim3((unsigned)((n + t - 1) / t)); };

  k_zero_f32<<<blk((long)NODE * OUTD, TB), TB, 0, stream>>>(outF, NODE * OUTD);
  k_zero_f32<<<blk(NODE, TB), TB, 0, stream>>>(cntE, NODE);
  k_zero_f32<<<blk(NODE, TB), TB, 0, stream>>>(cntR, NODE);
  k_zero_f32<<<blk(3L * 2 * BATCH, TB), TB, 0, stream>>>(rowsum, 3 * 2 * BATCH);
  k_zero_f32<<<1, 1, 0, stream>>>(accum, 1);

  k_sparse_acc<<<blk((long)TRIPLE * 4, TB), TB, 0, stream>>>(ent_adj, ent_adj + TRIPLE,
                                                             ent_emb, outF, 0, cntE, TRIPLE);
  k_sparse_acc<<<blk((long)TRIPLE * 4, TB), TB, 0, stream>>>(rel_adj, rel_adj + TRIPLE,
                                                             rel_emb, outF, 384, cntR, TRIPLE);
  k_mean_tanh<<<blk((long)NODE * DIM, TB), TB, 0, stream>>>(outF, 0, cntE);
  k_mean_tanh<<<blk((long)NODE * DIM, TB), TB, 0, stream>>>(outF, 384, cntR);

  k_rel_stats<<<blk((long)RELSZ * 32, TB), TB, 0, stream>>>(rel_emb, attn_e, attn_r, relnorm, dots);
  k_edge_scale<<<blk(TRIPLE, TB), TB, 0, stream>>>(r_index + TRIPLE, r_val, relnorm, dots, sE, att4);

  for (int p = 0; p < 2; ++p) {
    for (int l = 0; l < 2; ++l) {
      int phase = p * 2 + l;
      int inoff = p * 384 + l * 128;
      int outoff = inoff + 128;
      k_seg_init<<<blk(NODE, TB), TB, 0, stream>>>(segmaxU, segsum);
      k_seg_max<<<blk(TRIPLE, TB), TB, 0, stream>>>(adj_list, att4, phase, segmaxU);
      k_seg_exp<<<blk(TRIPLE, TB), TB, 0, stream>>>(adj_list, att4, phase, segmaxU, segsum, wbuf);
      k_aggregate<<<blk((long)TRIPLE * 32, TB), TB, 0, stream>>>(adj_list, r_index + TRIPLE,
                                                                 sE, wbuf, rel_emb, outF,
                                                                 inoff, outoff);
      k_soft_finish<<<blk((long)NODE * DIM, TB), TB, 0, stream>>>(outF, outoff, segsum);
    }
  }

  k_embsq<<<blk((long)NODE * 32, TB), TB, 0, stream>>>(outF, embsq);
  k_tobf16<<<blk((long)NODE * OUTD, TB), TB, 0, stream>>>(outF, embB, NODE * OUTD);
  k_pos<<<blk((long)BATCH * 32, TB), TB, 0, stream>>>(pairs, outF, posb);

  dim3 g(BATCH / 32, 2, CSPLIT);
  k_gemm<0><<<g, 256, 0, stream>>>(pairs, embB, embsq, posb,
                                   rowsum, rowsq, rowmaxU, rowmaxF, rowils, rowsumexp);
  k_rowfin<<<blk(2L * BATCH, TB), TB, 0, stream>>>(rowsum, rowsq, rowmaxU,
                                                   rowmu, rowils, rowmaxF, rowsumexp);
  k_gemm<1><<<g, 256, 0, stream>>>(pairs, embB, embsq, posb,
                                   rowsum, rowsq, rowmaxU, rowmaxF, rowils, rowsumexp);
  k_rowloss<<<blk(2L * BATCH, TB), TB, 0, stream>>>(rowmu, rowils, rowmaxF, rowsumexp, accum);
  k_final<<<1, 1, 0, stream>>>(accum, (float*)d_out);
}
